// DynamicBind_63840393888317
// MI455X (gfx1250) — compile-verified, matrix operand order checked by reading, not run on hardware
//
#include <hip/hip_runtime.h>
#include <math.h>

#define DIMD 128
#define NB   8
#define NL   128
#define NT   512
#define NGNN 3
#define KEMB 64   // embedding K padded 54 -> 64

typedef __attribute__((ext_vector_type(16))) _Float16 v16h;
typedef __attribute__((ext_vector_type(8)))  float    v8f;

// ---------------------------------------------------------------------------
// Block reductions (wave32 shuffles + LDS across waves)
// ---------------------------------------------------------------------------
__device__ __forceinline__ float block_reduce_sum(float v) {
  __shared__ float red[8];
  for (int off = 16; off; off >>= 1) v += __shfl_down(v, off, 32);
  const int lane = threadIdx.x & 31, w = threadIdx.x >> 5;
  const int nw = (blockDim.x + 31) >> 5;
  if (lane == 0) red[w] = v;
  __syncthreads();
  float r = 0.f;
  if ((int)threadIdx.x < nw) r = red[threadIdx.x];
  if (w == 0) for (int off = 16; off; off >>= 1) r += __shfl_down(r, off, 32);
  if (threadIdx.x == 0) red[0] = r;
  __syncthreads();
  r = red[0];
  __syncthreads();
  return r;
}

__device__ __forceinline__ float block_reduce_max(float v) {
  __shared__ float red[8];
  for (int off = 16; off; off >>= 1) v = fmaxf(v, __shfl_down(v, off, 32));
  const int lane = threadIdx.x & 31, w = threadIdx.x >> 5;
  const int nw = (blockDim.x + 31) >> 5;
  if (lane == 0) red[w] = v;
  __syncthreads();
  float r = -3.0e38f;
  if ((int)threadIdx.x < nw) r = red[threadIdx.x];
  if (w == 0) for (int off = 16; off; off >>= 1) r = fmaxf(r, __shfl_down(r, off, 32));
  if (threadIdx.x == 0) red[0] = r;
  __syncthreads();
  r = red[0];
  __syncthreads();
  return r;
}

// ---------------------------------------------------------------------------
// Zero-pad [rows,54] -> [rows,64] so GEMM K is always a multiple of 32.
// ---------------------------------------------------------------------------
__global__ __launch_bounds__(256) void pad54_kernel(
    const float* __restrict__ src, float* __restrict__ dst, long rows)
{
  const long i = (long)blockIdx.x * 256 + threadIdx.x;
  if (i >= rows * KEMB) return;
  const long r = i >> 6;
  const int  c = (int)(i & 63);
  dst[i] = (c < 54) ? src[r * 54 + c] : 0.f;
}

// ---------------------------------------------------------------------------
// WMMA GEMM: C[b] = A[b] * op(B[b]) (+bias row), f32 in/out, f16 MACs.
// One wave -> 16x64 C panel (4 accumulators, A fragment reused 4x).
// Requirements: M % 16 == 0, N % 64 == 0, K % 32 == 0 (no bounds checks).
// A fragment (16x32 f16): lane m=l&15, half=l>>4; vgpr-pair j holds
//   K = (j<4 ? 2j+8*half : 16+2(j-4)+8*half).  Pairs are K-contiguous ->
//   loaded as float2 (b64).  B mirrors with N; transB path also uses float2.
// D tile: vgpr r -> rows r (lanes 0-15) and r+8 (lanes 16-31), col = lane&15.
// ---------------------------------------------------------------------------
template<bool TRANSB>
__global__ __launch_bounds__(256) void gemm_wmma(
    const float* __restrict__ A, int lda, long sA,
    const float* __restrict__ B, int ldb, long sB,
    float* __restrict__ C, int ldc, long sC,
    int M, int N, int K, const float* __restrict__ bias)
{
  const int wave = threadIdx.x >> 5;
  const int lane = threadIdx.x & 31;
  const int tiles_n = N >> 6;                 // 64-wide N panels
  const int tiles   = (M >> 4) * tiles_n;
  const int tile    = blockIdx.x * 8 + wave;
  if (tile >= tiles) return;                  // wave-uniform
  const int tm = tile / tiles_n;
  const int tn = tile % tiles_n;
  const float* Ab = A + (long)blockIdx.y * sA;
  const float* Bb = B + (long)blockIdx.y * sB;
  float*       Cb = C + (long)blockIdx.y * sC;
  const int half  = lane >> 4;
  const int l15   = lane & 15;
  const int mrow  = (tm << 4) + l15;
  const int ncol0 = (tn << 6) + l15;

  const float* arow = Ab + (long)mrow * lda;

  int kp[8];
#pragma unroll
  for (int j = 0; j < 8; ++j)
    kp[j] = (j < 4) ? (j * 2 + half * 8) : (16 + (j - 4) * 2 + half * 8);

  v8f acc0 = {0.f,0.f,0.f,0.f,0.f,0.f,0.f,0.f};
  v8f acc1 = acc0, acc2 = acc0, acc3 = acc0;

  for (int k0 = 0; k0 < K; k0 += 32) {
    __builtin_prefetch((const void*)(arow + k0 + 32), 0, 1);
    v16h af;
#pragma unroll
    for (int j = 0; j < 8; ++j) {
      const float2 av = *(const float2*)(arow + k0 + kp[j]);
      af[2 * j]     = (_Float16)av.x;
      af[2 * j + 1] = (_Float16)av.y;
    }
#pragma unroll
    for (int s = 0; s < 4; ++s) {
      v16h bf;
      if (TRANSB) {
        const float* brow = Bb + (long)(ncol0 + 16 * s) * ldb;
#pragma unroll
        for (int j = 0; j < 8; ++j) {
          const float2 bv = *(const float2*)(brow + k0 + kp[j]);
          bf[2 * j]     = (_Float16)bv.x;
          bf[2 * j + 1] = (_Float16)bv.y;
        }
      } else {
        const float* bcol = Bb + ncol0 + 16 * s;
#pragma unroll
        for (int j = 0; j < 8; ++j) {
          bf[2 * j]     = (_Float16)bcol[(long)(k0 + kp[j])     * ldb];
          bf[2 * j + 1] = (_Float16)bcol[(long)(k0 + kp[j] + 1) * ldb];
        }
      }
      if      (s == 0) acc0 = __builtin_amdgcn_wmma_f32_16x16x32_f16(false, af, false, bf, (short)0, acc0, false, false);
      else if (s == 1) acc1 = __builtin_amdgcn_wmma_f32_16x16x32_f16(false, af, false, bf, (short)0, acc1, false, false);
      else if (s == 2) acc2 = __builtin_amdgcn_wmma_f32_16x16x32_f16(false, af, false, bf, (short)0, acc2, false, false);
      else             acc3 = __builtin_amdgcn_wmma_f32_16x16x32_f16(false, af, false, bf, (short)0, acc3, false, false);
    }
  }

#pragma unroll
  for (int s = 0; s < 4; ++s) {
    const v8f ac = (s == 0) ? acc0 : (s == 1) ? acc1 : (s == 2) ? acc2 : acc3;
    const int nc = ncol0 + 16 * s;
    const float bv = bias ? bias[nc] : 0.f;
#pragma unroll
    for (int r = 0; r < 8; ++r)
      Cb[(long)((tm << 4) + r + half * 8) * ldc + nc] = ac[r] + bv;
  }
}

// ---------------------------------------------------------------------------
// Row softmax for attention: att[b,j,k] = softmax_k(mask(e[b,j,k]+e[b,k,j]))*adj
// One block per (b,j) row.
// ---------------------------------------------------------------------------
__global__ __launch_bounds__(256) void att_softmax_kernel(
    const float* __restrict__ e, const float* __restrict__ adj,
    float* __restrict__ att, int N)
{
  __shared__ float srow[512];
  const int row = blockIdx.x;
  const int b = row / N, j = row % N;
  const float* eb   = e   + (long)b * N * N;
  const float* arow = adj + (long)b * N * N + (long)j * N;
  float lmax = -3.0e38f;
  for (int k = threadIdx.x; k < N; k += blockDim.x) {
    const float s = (arow[k] > 0.f) ? (eb[(long)j * N + k] + eb[(long)k * N + j])
                                    : -9.0e15f;
    srow[k] = s;
    lmax = fmaxf(lmax, s);
  }
  const float m = block_reduce_max(lmax);
  float lsum = 0.f;
  for (int k = threadIdx.x; k < N; k += blockDim.x) {
    const float ex = __expf(srow[k] - m);
    srow[k] = ex;
    lsum += ex;
  }
  const float ssum = block_reduce_sum(lsum);
  const float inv  = 1.0f / ssum;
  float* orow = att + (long)b * N * N + (long)j * N;
  for (int k = threadIdx.x; k < N; k += blockDim.x)
    orow[k] = srow[k] * inv * arow[k];
}

// ---------------------------------------------------------------------------
// Gated residual: hp = relu(hp_raw); coeff = sigmoid([x,hp]@gW + gb);
// x = coeff*x + (1-coeff)*hp.  One block (128 threads) per (b,node).
// ---------------------------------------------------------------------------
__global__ __launch_bounds__(128) void gate_kernel(
    float* __restrict__ x, const float* __restrict__ hp_raw,
    const float* __restrict__ gW, const float* __restrict__ gb)
{
  const long row = blockIdx.x;
  const int  t   = threadIdx.x;
  const float xv = x[row * DIMD + t];
  const float hv = fmaxf(hp_raw[row * DIMD + t], 0.f);
  const float s  = block_reduce_sum(xv * gW[t] + hv * gW[DIMD + t]);
  const float coeff = 1.0f / (1.0f + __expf(-(s + gb[0])));
  x[row * DIMD + t] = coeff * xv + (1.0f - coeff) * hv;
}

// ---------------------------------------------------------------------------
// Fused pair stage: per (b, 16x16 pair tile) stage ligK/tgtK rows + W2 in LDS,
// evaluate both pair MLPs (relu dot), distances, vdW + ramp energies,
// block-reduce the 4 energy channels and atomically accumulate per batch.
// ---------------------------------------------------------------------------
__global__ __launch_bounds__(256) void pair_energy_kernel(
    const float* __restrict__ lKdc, const float* __restrict__ lKv,
    const float* __restrict__ tKdc, const float* __restrict__ tKv,
    const float* __restrict__ dcW2, const float* __restrict__ dcb2,
    const float* __restrict__ vW2,  const float* __restrict__ vb2,
    const float* __restrict__ ligand_pos, const float* __restrict__ target_pos,
    const float* __restrict__ lig_r, const float* __restrict__ tgt_r,
    const float* __restrict__ lig_valid, const float* __restrict__ tgt_valid,
    const float* __restrict__ ii, float* __restrict__ acc)
{
  __shared__ float sLdc[16][DIMD];
  __shared__ float sLv [16][DIMD];
  __shared__ float sTdc[16][DIMD];
  __shared__ float sTv [16][DIMD];
  __shared__ float sW2dc[DIMD], sW2v[DIMD];
  __shared__ float sLpos[16][3], sTpos[16][3];
  __shared__ float sLr[16], sTr[16], sLval[16], sTval[16];

  const int b    = blockIdx.y;
  const int tile = blockIdx.x;                 // (L/16)*(T/16) tiles
  const int tt0  = (tile % (NT / 16)) * 16;
  const int tl0  = (tile / (NT / 16)) * 16;
  const int tid  = threadIdx.x;

  for (int idx = tid; idx < 16 * DIMD; idx += 256) {
    const int r = idx >> 7, k = idx & (DIMD - 1);
    const long lrow = (long)(b * NL + tl0 + r) * DIMD + k;
    const long trow = (long)(b * NT + tt0 + r) * DIMD + k;
    sLdc[r][k] = lKdc[lrow];
    sLv [r][k] = lKv [lrow];
    sTdc[r][k] = tKdc[trow];
    sTv [r][k] = tKv [trow];
  }
  for (int k = tid; k < DIMD; k += 256) { sW2dc[k] = dcW2[k]; sW2v[k] = vW2[k]; }
  if (tid < 16) {
    const int l = tl0 + tid, t = tt0 + tid;
    sLpos[tid][0] = ligand_pos[(b * NL + l) * 3 + 0];
    sLpos[tid][1] = ligand_pos[(b * NL + l) * 3 + 1];
    sLpos[tid][2] = ligand_pos[(b * NL + l) * 3 + 2];
    sTpos[tid][0] = target_pos[(b * NT + t) * 3 + 0];
    sTpos[tid][1] = target_pos[(b * NT + t) * 3 + 1];
    sTpos[tid][2] = target_pos[(b * NT + t) * 3 + 2];
    sLr[tid]   = lig_r[b * NL + l];
    sTr[tid]   = tgt_r[b * NT + t];
    sLval[tid] = lig_valid[b * NL + l];
    sTval[tid] = tgt_valid[b * NT + t];
  }
  __syncthreads();

  const int tl = tid >> 4, tt = tid & 15;
  float sdc = 0.f, sv = 0.f;
#pragma unroll 4
  for (int k = 0; k < DIMD; ++k) {
    const float p1 = sLdc[tl][k] + sTdc[tt][k];
    sdc += fmaxf(p1, 0.f) * sW2dc[k];
    const float p2 = sLv[tl][k] + sTv[tt][k];
    sv  += fmaxf(p2, 0.f) * sW2v[k];
  }
  sdc += dcb2[0];
  sv  += vb2[0];

  const float dx = sLpos[tl][0] - sTpos[tt][0];
  const float dy = sLpos[tl][1] - sTpos[tt][1];
  const float dz = sLpos[tl][2] - sTpos[tt][2];
  float dm = sqrtf(dx * dx + dy * dy + dz * dz + 1e-10f);
  if (dm < 0.5f) dm = 1.0e10f;

  const float vdws = sLr[tl] + sTr[tt] + tanhf(sdc) * 0.2f;   // DEV_VDW_RADIUS
  const float dm0  = (vdws < 1e-4f) ? 1.0f : vdws;
  const float ratio = dm0 / dm;
  const float r2 = ratio * ratio, r6 = r2 * r2 * r2, r12 = r6 * r6;
  const float valid  = sLval[tl] * sTval[tt];
  const float energy = fminf(r12 - 2.0f * r6, 100.0f) * valid;
  float Aw = 1.0f / (1.0f + __expf(-sv));
  Aw = Aw * (0.0356f - 0.0178f) + 0.0178f;                    // MAX/MIN_VDW
  const float vdw_c = Aw * energy;

  const float d     = dm - vdws;
  const float ramp  = fminf(fmaxf(d * (-1.0f / 0.7f), 0.f), 1.f);
  const float ramp2 = fminf(fmaxf(-d + 1.5f, 0.f), 1.f);
  const int l = tl0 + tl, t = tt0 + tt;
  const long iib = ((long)(b * 3 + 0) * NL + l) * NT + t;
  const float hb = ramp  * ii[iib];
  const float mt = ramp  * ii[iib + (long)NL * NT];
  const float hy = ramp2 * ii[iib + 2L * NL * NT];

  const float rv = block_reduce_sum(vdw_c);
  const float rh = block_reduce_sum(hb);
  const float rm = block_reduce_sum(mt);
  const float ry = block_reduce_sum(hy);
  if (tid == 0) {
    atomicAdd(&acc[b * 4 + 0], rv);
    atomicAdd(&acc[b * 4 + 1], rh);
    atomicAdd(&acc[b * 4 + 2], rm);
    atomicAdd(&acc[b * 4 + 3], ry);
  }
}

// ---------------------------------------------------------------------------
// Apply coefficient signs and rotor penalty -> energies [B,4]
// ---------------------------------------------------------------------------
__global__ void finalize_kernel(const float* __restrict__ acc,
                                const float* __restrict__ rotor,
                                const float* __restrict__ hbond_c,
                                const float* __restrict__ metal_c,
                                const float* __restrict__ hydro_c,
                                const float* __restrict__ rotor_c,
                                float* __restrict__ out)
{
  const int i = threadIdx.x;
  if (i >= NB * 4) return;
  const int b = i >> 2, c = i & 3;
  float v = acc[i];
  if      (c == 1) v *= -(hbond_c[0] * hbond_c[0]);
  else if (c == 2) v *= -(metal_c[0] * metal_c[0]);
  else if (c == 3) v *= -(hydro_c[0] * hydro_c[0]);
  const float denom = 1.0f + rotor_c[0] * rotor_c[0] * rotor[b];
  out[i] = v / denom;
}

// ---------------------------------------------------------------------------
extern "C" void kernel_launch(void* const* d_in, const int* in_sizes, int n_in,
                              void* d_out, int out_size, void* d_ws, size_t ws_size,
                              hipStream_t stream)
{
  const float* ligand_h    = (const float*)d_in[0];
  const float* target_h    = (const float*)d_in[1];
  const float* ligand_adj  = (const float*)d_in[2];
  const float* target_adj  = (const float*)d_in[3];
  const float* inter_ind   = (const float*)d_in[4];
  const float* ligand_pos  = (const float*)d_in[5];
  const float* target_pos  = (const float*)d_in[6];
  const float* rotor       = (const float*)d_in[7];
  const float* lig_vdw_r   = (const float*)d_in[8];
  const float* tgt_vdw_r   = (const float*)d_in[9];
  const float* lig_valid   = (const float*)d_in[10];
  const float* tgt_valid   = (const float*)d_in[11];
  const float* emb_W       = (const float*)d_in[12];
  const float* gat_W       = (const float*)d_in[13];
  const float* gat_Wb      = (const float*)d_in[14];
  const float* gat_A       = (const float*)d_in[15];
  const float* gat_gW      = (const float*)d_in[16];
  const float* gat_gb      = (const float*)d_in[17];
  const float* vdwA_W1     = (const float*)d_in[18];
  const float* vdwA_b1     = (const float*)d_in[19];
  const float* vdwA_W2     = (const float*)d_in[20];
  const float* vdwA_b2     = (const float*)d_in[21];
  const float* dc_W1       = (const float*)d_in[22];
  const float* dc_b1       = (const float*)d_in[23];
  const float* dc_W2       = (const float*)d_in[24];
  const float* dc_b2       = (const float*)d_in[25];
  const float* hbond_c     = (const float*)d_in[26];
  const float* metal_c     = (const float*)d_in[27];
  const float* hydro_c     = (const float*)d_in[28];
  const float* rotor_c     = (const float*)d_in[29];

  // ---- workspace layout (floats), ~32 MB total ----
  float* p = (float*)d_ws;
  float* lig    = p; p += (long)NB * NL * DIMD;
  float* tgt    = p; p += (long)NB * NT * DIMD;
  float* h_l    = p; p += (long)NB * NL * DIMD;
  float* hA_l   = p; p += (long)NB * NL * DIMD;
  float* e_l    = p; p += (long)NB * NL * NL;
  float* att_l  = p; p += (long)NB * NL * NL;
  float* hp_l   = p; p += (long)NB * NL * DIMD;
  float* h_t    = p; p += (long)NB * NT * DIMD;
  float* hA_t   = p; p += (long)NB * NT * DIMD;
  float* e_t    = p; p += (long)NB * NT * NT;
  float* att_t  = p; p += (long)NB * NT * NT;
  float* hp_t   = p; p += (long)NB * NT * DIMD;
  float* lKdc   = p; p += (long)NB * NL * DIMD;
  float* lKv    = p; p += (long)NB * NL * DIMD;
  float* tKdc   = p; p += (long)NB * NT * DIMD;
  float* tKv    = p; p += (long)NB * NT * DIMD;
  float* lig_hp = p; p += (long)NB * NL * KEMB;   // padded embeddings inputs
  float* tgt_hp = p; p += (long)NB * NT * KEMB;
  float* acc    = p; p += NB * 4;

  hipMemsetAsync(acc, 0, NB * 4 * sizeof(float), stream);

  auto gemm = [&](const float* A, int lda, long sA,
                  const float* B, int ldb, long sB,
                  float* C, int ldc, long sC,
                  int M, int N, int K, const float* bias, bool transB, int batch) {
    const int tiles = (M >> 4) * (N >> 6);
    dim3 grid((tiles + 7) / 8, batch);
    if (transB)
      gemm_wmma<true ><<<grid, 256, 0, stream>>>(A, lda, sA, B, ldb, sB,
                                                 C, ldc, sC, M, N, K, bias);
    else
      gemm_wmma<false><<<grid, 256, 0, stream>>>(A, lda, sA, B, ldb, sB,
                                                 C, ldc, sC, M, N, K, bias);
  };

  // ---- pad node features to K=64, then embed: x = h_pad @ emb_W ----
  {
    const long nl = (long)NB * NL * KEMB, nt = (long)NB * NT * KEMB;
    pad54_kernel<<<(int)((nl + 255) / 256), 256, 0, stream>>>(ligand_h, lig_hp, NB * NL);
    pad54_kernel<<<(int)((nt + 255) / 256), 256, 0, stream>>>(target_h, tgt_hp, NB * NT);
  }
  // emb_W is [54,128]; padded rows 54..63 of a [64,128] view read past it but
  // are multiplied by the zero-padded A columns. To stay in-bounds, treat B
  // with TRANSB=false and ldb=DIMD reading rows k<64: rows 54..63 would be OOB.
  // Instead pad emb_W too: reuse tail of workspace for a [64,128] copy.
  float* embWp = p; p += (long)KEMB * DIMD;
  {
    // pad kernel reuse: emb_W^T view is not needed; do a dedicated small copy
    // using pad semantics along rows: dst[r*128+c] = (r<54) ? src[r*128+c] : 0
    // Implemented with pad54_kernel layout trick is wrong; use memset + copy.
    hipMemsetAsync(embWp, 0, (size_t)KEMB * DIMD * sizeof(float), stream);
    hipMemcpyAsync(embWp, emb_W, (size_t)54 * DIMD * sizeof(float),
                   hipMemcpyDeviceToDevice, stream);
  }
  gemm(lig_hp, KEMB, 0, embWp, DIMD, 0, lig, DIMD, 0, NB * NL, DIMD, KEMB, nullptr, false, 1);
  gemm(tgt_hp, KEMB, 0, embWp, DIMD, 0, tgt, DIMD, 0, NB * NT, DIMD, KEMB, nullptr, false, 1);

  // ---- 3 GatedGAT layers on both graphs ----
  for (int i = 0; i < NGNN; ++i) {
    const float* Wi  = gat_W  + (long)i * DIMD * DIMD;
    const float* Wbi = gat_Wb + i * DIMD;
    const float* Ai  = gat_A  + (long)i * DIMD * DIMD;
    const float* gWi = gat_gW + i * 2 * DIMD;
    const float* gbi = gat_gb + i;

    // ligand graph (N nodes = 128)
    gemm(lig, DIMD, 0, Wi, DIMD, 0, h_l,  DIMD, 0, NB * NL, DIMD, DIMD, Wbi, false, 1);
    gemm(h_l, DIMD, 0, Ai, DIMD, 0, hA_l, DIMD, 0, NB * NL, DIMD, DIMD, nullptr, false, 1);
    gemm(hA_l, DIMD, (long)NL * DIMD, h_l, DIMD, (long)NL * DIMD,
         e_l, NL, (long)NL * NL, NL, NL, DIMD, nullptr, true, NB);       // e = hA . h^T
    att_softmax_kernel<<<NB * NL, 256, 0, stream>>>(e_l, ligand_adj, att_l, NL);
    gemm(att_l, NL, (long)NL * NL, h_l, DIMD, (long)NL * DIMD,
         hp_l, DIMD, (long)NL * DIMD, NL, DIMD, NL, nullptr, false, NB); // hp = att . h
    gate_kernel<<<NB * NL, 128, 0, stream>>>(lig, hp_l, gWi, gbi);

    // target graph (N nodes = 512)
    gemm(tgt, DIMD, 0, Wi, DIMD, 0, h_t,  DIMD, 0, NB * NT, DIMD, DIMD, Wbi, false, 1);
    gemm(h_t, DIMD, 0, Ai, DIMD, 0, hA_t, DIMD, 0, NB * NT, DIMD, DIMD, nullptr, false, 1);
    gemm(hA_t, DIMD, (long)NT * DIMD, h_t, DIMD, (long)NT * DIMD,
         e_t, NT, (long)NT * NT, NT, NT, DIMD, nullptr, true, NB);
    att_softmax_kernel<<<NB * NT, 256, 0, stream>>>(e_t, target_adj, att_t, NT);
    gemm(att_t, NT, (long)NT * NT, h_t, DIMD, (long)NT * DIMD,
         hp_t, DIMD, (long)NT * DIMD, NT, DIMD, NT, nullptr, false, NB);
    gate_kernel<<<NB * NT, 128, 0, stream>>>(tgt, hp_t, gWi, gbi);
  }

  // ---- pair-MLP half-contractions (b1 folded into ligand half) ----
  gemm(lig, DIMD, 0, dc_W1,               DIMD, 0, lKdc, DIMD, 0, NB * NL, DIMD, DIMD, dc_b1,   false, 1);
  gemm(tgt, DIMD, 0, dc_W1 + DIMD * DIMD, DIMD, 0, tKdc, DIMD, 0, NB * NT, DIMD, DIMD, nullptr, false, 1);
  gemm(lig, DIMD, 0, vdwA_W1,               DIMD, 0, lKv, DIMD, 0, NB * NL, DIMD, DIMD, vdwA_b1, false, 1);
  gemm(tgt, DIMD, 0, vdwA_W1 + DIMD * DIMD, DIMD, 0, tKv, DIMD, 0, NB * NT, DIMD, DIMD, nullptr, false, 1);

  // ---- fused pair energies + reductions ----
  dim3 pg((NL / 16) * (NT / 16), NB);
  pair_energy_kernel<<<pg, 256, 0, stream>>>(
      lKdc, lKv, tKdc, tKv, dc_W2, dc_b2, vdwA_W2, vdwA_b2,
      ligand_pos, target_pos, lig_vdw_r, tgt_vdw_r, lig_valid, tgt_valid,
      inter_ind, acc);

  // ---- rotor penalty + signs ----
  finalize_kernel<<<1, 32, 0, stream>>>(acc, rotor, hbond_c, metal_c, hydro_c,
                                        rotor_c, (float*)d_out);
}
